// GraphAttention_3161095930276
// MI455X (gfx1250) — matrix operand order repaired and checked by reading, not compiled, not run on hardware
//
#include <hip/hip_runtime.h>
#include <hip/hip_bf16.h>
#include <math.h>

#define GA_N     12288
#define GA_FIN   256
#define GA_FOUT  64
#define GA_ALPHA 0.2f
#define GA_NEGBIG (-9.0e15f)
#define GA_RB    (GA_N / 16)   /* 768 row blocks */

typedef __attribute__((ext_vector_type(16))) __bf16 v16bf;
typedef __attribute__((ext_vector_type(8)))  __bf16 v8bf;
typedef __attribute__((ext_vector_type(8)))  float  v8f;

// ---------------------------------------------------------------------------
// Kernel 1: h = input @ W (fp32), emit hbfT[64][N] (bf16, transposed -> WMMA B
// operand) and f1 = h@a1, f2 = h@a2 (fp32).
// ---------------------------------------------------------------------------
__global__ void __launch_bounds__(256) ga_proj_kernel(
    const float* __restrict__ input, const float* __restrict__ W,
    const float* __restrict__ a1, const float* __restrict__ a2,
    __bf16* __restrict__ hbfT, float* __restrict__ f1, float* __restrict__ f2)
{
  const int tid = threadIdx.x;
  const int r   = tid >> 4;
  const int q   = tid & 15;
  const int row = blockIdx.x * 16 + r;
  const int c   = q * 4;

  const float* in = input + (size_t)row * GA_FIN;
  float acc0 = 0.f, acc1 = 0.f, acc2 = 0.f, acc3 = 0.f;
#pragma unroll 4
  for (int k = 0; k < GA_FIN; ++k) {
    const float x = in[k];
    const float4 wv = *reinterpret_cast<const float4*>(W + (size_t)k * GA_FOUT + c);
    acc0 = fmaf(x, wv.x, acc0);
    acc1 = fmaf(x, wv.y, acc1);
    acc2 = fmaf(x, wv.z, acc2);
    acc3 = fmaf(x, wv.w, acc3);
  }

  hbfT[(size_t)(c + 0) * GA_N + row] = (__bf16)acc0;
  hbfT[(size_t)(c + 1) * GA_N + row] = (__bf16)acc1;
  hbfT[(size_t)(c + 2) * GA_N + row] = (__bf16)acc2;
  hbfT[(size_t)(c + 3) * GA_N + row] = (__bf16)acc3;

  const float4 a1v = *reinterpret_cast<const float4*>(a1 + c);
  const float4 a2v = *reinterpret_cast<const float4*>(a2 + c);
  float s1 = acc0 * a1v.x + acc1 * a1v.y + acc2 * a1v.z + acc3 * a1v.w;
  float s2 = acc0 * a2v.x + acc1 * a2v.y + acc2 * a2v.z + acc3 * a2v.w;
#pragma unroll
  for (int off = 8; off > 0; off >>= 1) {
    s1 += __shfl_xor(s1, off, 32);
    s2 += __shfl_xor(s2, off, 32);
  }
  if (q == 0) { f1[row] = s1; f2[row] = s2; }
}

// ---------------------------------------------------------------------------
// Kernel 1b: gmax = max_j f2[j]  (single block)
// ---------------------------------------------------------------------------
__global__ void __launch_bounds__(256) ga_gmax_kernel(
    const float* __restrict__ f2, float* __restrict__ gmaxOut)
{
  float m = -3.0e38f;
  for (int i = threadIdx.x; i < GA_N; i += 256) m = fmaxf(m, f2[i]);
#pragma unroll
  for (int off = 16; off > 0; off >>= 1) m = fmaxf(m, __shfl_xor(m, off, 32));
  __shared__ float smax[8];
  if ((threadIdx.x & 31) == 0) smax[threadIdx.x >> 5] = m;
  __syncthreads();
  if (threadIdx.x == 0) {
    float r = smax[0];
#pragma unroll
    for (int i = 1; i < 8; ++i) r = fmaxf(r, smax[i]);
    gmaxOut[0] = r;
  }
}

// ---------------------------------------------------------------------------
// Kernel 2: fused masked-attention, one wave per (row-block, column-split).
// Static softmax bound M_i = leaky(f1[i] + gmax)  =>  no rescaling, no per-tile
// shuffles. P@H via v_wmma_f32_16x16x32_bf16 (4 n-tiles of 16).
// ---------------------------------------------------------------------------
__device__ __forceinline__ float4 ga_edge(float4 f2v, int4 av, float f1m) {
  float4 e;
  e.x = f1m + f2v.x; e.x = e.x >= 0.f ? e.x : GA_ALPHA * e.x; if (av.x <= 0) e.x = GA_NEGBIG;
  e.y = f1m + f2v.y; e.y = e.y >= 0.f ? e.y : GA_ALPHA * e.y; if (av.y <= 0) e.y = GA_NEGBIG;
  e.z = f1m + f2v.z; e.z = e.z >= 0.f ? e.z : GA_ALPHA * e.z; if (av.z <= 0) e.z = GA_NEGBIG;
  e.w = f1m + f2v.w; e.w = e.w >= 0.f ? e.w : GA_ALPHA * e.w; if (av.w <= 0) e.w = GA_NEGBIG;
  return e;
}

__global__ void __launch_bounds__(128) ga_attn_partial(
    const int* __restrict__ adj, const float* __restrict__ f1,
    const float* __restrict__ f2, const float* __restrict__ gmaxp,
    const __bf16* __restrict__ hbfT,
    float* __restrict__ pl, float* __restrict__ pacc, int nsplit)
{
  const int lane = threadIdx.x & 31;
  const int gw   = blockIdx.x * (blockDim.x >> 5) + (threadIdx.x >> 5);
  const int rb   = gw / nsplit;
  const int sp   = gw - rb * nsplit;
  const int r0   = rb << 4;
  const int cchunk = GA_N / nsplit;
  const int c0   = sp * cchunk;
  const int mrow = lane & 15;      // A row ownership: lanes m and m+16 share row m; C column m
  const int half = lane >> 4;

  const float f1m = f1[r0 + mrow];
  // static per-row softmax bound: M >= every e in this row (leaky is monotone)
  float M = f1m + gmaxp[0];
  M = (M >= 0.f) ? M : GA_ALPHA * M;

  const int* adjRow = adj + (size_t)(r0 + mrow) * GA_N;

  // B operand base pointers: lane owns column n = lane&15, K = 8*half + e (e<8), 16 + 8*half + (e-8) (e>=8)
  const __bf16* hb0 = hbfT + (size_t)( 0 + mrow) * GA_N + 8 * half;
  const __bf16* hb1 = hbfT + (size_t)(16 + mrow) * GA_N + 8 * half;
  const __bf16* hb2 = hbfT + (size_t)(32 + mrow) * GA_N + 8 * half;
  const __bf16* hb3 = hbfT + (size_t)(48 + mrow) * GA_N + 8 * half;

  float runL = 0.f;                // per-lane partial row sum; combined once at end
  v8f acc0 = {}, acc1 = {}, acc2 = {}, acc3 = {};

  for (int jt = c0; jt < c0 + cchunk; jt += 32) {
    // ---- issue all loads up front so the exp chain hides their latency ----
    const v8bf lo0 = *reinterpret_cast<const v8bf*>(hb0 + jt);
    const v8bf hi0 = *reinterpret_cast<const v8bf*>(hb0 + jt + 16);
    const v8bf lo1 = *reinterpret_cast<const v8bf*>(hb1 + jt);
    const v8bf hi1 = *reinterpret_cast<const v8bf*>(hb1 + jt + 16);
    const v8bf lo2 = *reinterpret_cast<const v8bf*>(hb2 + jt);
    const v8bf hi2 = *reinterpret_cast<const v8bf*>(hb2 + jt + 16);
    const v8bf lo3 = *reinterpret_cast<const v8bf*>(hb3 + jt);
    const v8bf hi3 = *reinterpret_cast<const v8bf*>(hb3 + jt + 16);

    const int j0 = jt + 8 * half;        // A element indices 0..7
    const int j1 = jt + 16 + 8 * half;   // A element indices 8..15
    __builtin_prefetch(adjRow + jt + 512, 0, 0);
    const int4 A0 = *reinterpret_cast<const int4*>(adjRow + j0);
    const int4 A1 = *reinterpret_cast<const int4*>(adjRow + j0 + 4);
    const int4 A2 = *reinterpret_cast<const int4*>(adjRow + j1);
    const int4 A3 = *reinterpret_cast<const int4*>(adjRow + j1 + 4);
    const float4 F0 = *reinterpret_cast<const float4*>(f2 + j0);
    const float4 F1 = *reinterpret_cast<const float4*>(f2 + j0 + 4);
    const float4 F2 = *reinterpret_cast<const float4*>(f2 + j1);
    const float4 F3 = *reinterpret_cast<const float4*>(f2 + j1 + 4);

    const float4 e0 = ga_edge(F0, A0, f1m);
    const float4 e1 = ga_edge(F1, A1, f1m);
    const float4 e2 = ga_edge(F2, A2, f1m);
    const float4 e3 = ga_edge(F3, A3, f1m);

    // p = exp(e - M) in WMMA-A bf16 layout; accumulate per-lane sum
    v16bf av;
    float psum = 0.f;
#define GA_P(EV, B)                                                          \
    { float p;                                                               \
      p = __expf(EV.x - M); psum += p; av[(B) + 0] = (__bf16)p;              \
      p = __expf(EV.y - M); psum += p; av[(B) + 1] = (__bf16)p;              \
      p = __expf(EV.z - M); psum += p; av[(B) + 2] = (__bf16)p;              \
      p = __expf(EV.w - M); psum += p; av[(B) + 3] = (__bf16)p; }
    GA_P(e0, 0) GA_P(e1, 4) GA_P(e2, 8) GA_P(e3, 12)
#undef GA_P
    runL += psum;

    v16bf b0, b1, b2, b3;
#pragma unroll
    for (int ii = 0; ii < 8; ++ii) {
      b0[ii] = lo0[ii]; b0[8 + ii] = hi0[ii];
      b1[ii] = lo1[ii]; b1[8 + ii] = hi1[ii];
      b2[ii] = lo2[ii]; b2[8 + ii] = hi2[ii];
      b3[ii] = lo3[ii]; b3[8 + ii] = hi3[ii];
    }

    acc0 = __builtin_amdgcn_wmma_f32_16x16x32_bf16(false, av, false, b0, (short)0, acc0, false, false);
    acc1 = __builtin_amdgcn_wmma_f32_16x16x32_bf16(false, av, false, b1, (short)0, acc1, false, false);
    acc2 = __builtin_amdgcn_wmma_f32_16x16x32_bf16(false, av, false, b2, (short)0, acc2, false, false);
    acc3 = __builtin_amdgcn_wmma_f32_16x16x32_bf16(false, av, false, b3, (short)0, acc3, false, false);
  }

  // combine the two disjoint K-halves of each row once
  runL += __shfl_xor(runL, 16, 32);
  if (half == 0) pl[(gw << 4) + mrow] = runL;

  float* accBase = pacc + (size_t)gw * 16 * GA_FOUT;
#pragma unroll
  for (int rr = 0; rr < 8; ++rr) {
    const int R = rr + 8 * half;
    float* rowp = accBase + R * GA_FOUT + mrow;
    rowp[0]  = acc0[rr];
    rowp[16] = acc1[rr];
    rowp[32] = acc2[rr];
    rowp[48] = acc3[rr];
  }
}

// ---------------------------------------------------------------------------
// Kernel 3: merge split partials (shared M => plain sums), normalize, ELU.
// ---------------------------------------------------------------------------
__global__ void __launch_bounds__(256) ga_combine(
    const float* __restrict__ pl, const float* __restrict__ pacc,
    float* __restrict__ out, int nsplit)
{
  const int idx = blockIdx.x * blockDim.x + threadIdx.x;  // row*64 + c
  const int row = idx >> 6;
  const int c   = idx & 63;
  const int rb  = row >> 4;
  const int rl  = row & 15;

  float L = 0.f, V = 0.f;
  for (int s = 0; s < nsplit; ++s) {
    const int g = rb * nsplit + s;
    L += pl[(g << 4) + rl];
    V += pacc[((size_t)g * 16 + rl) * GA_FOUT + c];
  }
  const float hp = V / L;
  out[idx] = hp > 0.f ? hp : (__expf(hp) - 1.f);
}

// ---------------------------------------------------------------------------
extern "C" void kernel_launch(void* const* d_in, const int* in_sizes, int n_in,
                              void* d_out, int out_size, void* d_ws, size_t ws_size,
                              hipStream_t stream) {
  const float* input = (const float*)d_in[0];
  const int*   adj   = (const int*)d_in[1];
  const float* W     = (const float*)d_in[2];
  const float* a1    = (const float*)d_in[3];
  const float* a2    = (const float*)d_in[4];
  float* out = (float*)d_out;

  // workspace carve-out (256 B aligned slices)
  char* ws = (char*)d_ws;
  size_t off = 0;
  auto carve = [&](size_t bytes) -> void* {
    void* p = ws + off;
    off = (off + bytes + 255) & ~(size_t)255;
    return p;
  };
  __bf16* hbfT = (__bf16*)carve((size_t)GA_FOUT * GA_N * sizeof(__bf16));
  float*  f1   = (float*)carve((size_t)GA_N * sizeof(float));
  float*  f2   = (float*)carve((size_t)GA_N * sizeof(float));
  float*  gmax = (float*)carve(sizeof(float));

  // deterministic split-count choice from ws_size
  int nsplit = 16;
  while (nsplit > 1) {
    const size_t per = (size_t)GA_RB * nsplit;
    const size_t need = off + 2 * 256 +
                        per * 16 * sizeof(float) +           // pl
                        per * 16 * GA_FOUT * sizeof(float);  // pacc
    if (need <= ws_size) break;
    nsplit >>= 1;
  }
  float* pl   = (float*)carve((size_t)GA_RB * nsplit * 16 * sizeof(float));
  float* pacc = (float*)carve((size_t)GA_RB * nsplit * 16 * GA_FOUT * sizeof(float));

  ga_proj_kernel<<<GA_RB, 256, 0, stream>>>(input, W, a1, a2, hbfT, f1, f2);
  ga_gmax_kernel<<<1, 256, 0, stream>>>(f2, gmax);

  const int waves = GA_RB * nsplit;          // 768 * nsplit, divisible by 4
  ga_attn_partial<<<waves / 4, 128, 0, stream>>>(adj, f1, f2, gmax, hbfT, pl, pacc, nsplit);

  ga_combine<<<(GA_N * GA_FOUT) / 256, 256, 0, stream>>>(pl, pacc, out, nsplit);
}
